// TransformerBlock_39556648796776
// MI455X (gfx1250) — matrix operand order repaired and checked by reading, not compiled, not run on hardware
//
#include <hip/hip_runtime.h>
#include <hip/hip_bf16.h>

namespace {

constexpr int kD     = 1024;
constexpr int kHeads = 16;
constexpr int kDk    = 64;
constexpr int kDff   = 4096;
constexpr int kS     = 2048;
constexpr int kB     = 4;
constexpr int kRows  = kB * kS;   // 8192

typedef __bf16 bf16_t;
typedef __attribute__((ext_vector_type(16))) __bf16 v16bf;
typedef __attribute__((ext_vector_type(8)))  __bf16 v8bf;
typedef __attribute__((ext_vector_type(8)))  float  v8f;

__device__ __forceinline__ v8f wmma_bf16f32(v16bf a, v16bf b, v8f c) {
  // D(16x16 f32) = A(16x32 bf16) * B(32x16 bf16) + C
  return __builtin_amdgcn_wmma_f32_16x16x32_bf16(false, a, false, b, (short)0, c,
                                                 false, false);
}

// Async DMA: 16 bytes per lane, global -> LDS, tracked by ASYNCcnt.
// GV form: vdst = LDS byte offset (low 32 bits of flat LDS address), vaddr64.
__device__ __forceinline__ void async_copy16(unsigned lds_byte_off,
                                             const bf16_t* src) {
  asm volatile("global_load_async_to_lds_b128 %0, %1, off"
               :
               : "v"(lds_byte_off), "v"((unsigned long long)(uintptr_t)src)
               : "memory");
}

__device__ __forceinline__ void wait_async_all() {
  asm volatile("s_wait_asynccnt 0x0" ::: "memory");
}

// A fragment: 16 rows (M) x 32 (K), K contiguous, row stride ld (elems).
// Lane m<16: elems 0..15 = K 0..15 ; lane m+16: K 16..31.
__device__ __forceinline__ v16bf load_a_frag(const bf16_t* base, int ld) {
  const int lane = threadIdx.x & 31;
  const int r    = lane & 15;
  const int koff = (lane >> 4) << 4;  // 0 or 16
  const bf16_t* p = base + (size_t)r * ld + koff;
  v8bf lo = *reinterpret_cast<const v8bf*>(p);
  v8bf hi = *reinterpret_cast<const v8bf*>(p + 8);
  v16bf f;
#pragma unroll
  for (int i = 0; i < 8; ++i) { f[i] = lo[i]; f[i + 8] = hi[i]; }
  return f;
}

// B fragment: 32 (K) x 16 cols (N); source row-major over K with stride ld,
// row index = N column. Lane n<16: K 0..15 ; lane n+16: K 16..31.
__device__ __forceinline__ v16bf load_b_frag(const bf16_t* base, int ld) {
  const int lane = threadIdx.x & 31;
  const int n    = lane & 15;
  const int koff = (lane >> 4) << 4;  // 0 or 16
  return *reinterpret_cast<const v16bf*>(base + (size_t)n * ld + koff);
}

enum GemmEpi {
  EPI_BIAS_BF16 = 0,      // C_bf16[row*N+col] = acc + bias
  EPI_BIAS_BF16_T = 1,    // V^T layout (B, heads, dk, S) bf16
  EPI_BIAS_RES_F32 = 2,   // C_f32 = acc + bias + res
  EPI_BIAS_GELU_BF16 = 3  // C_bf16 = gelu(acc + bias)
};

// out[m,n] = sum_k A[m,k] * W[n,k]   (A: MxK bf16 rm, W: NxK bf16 rm)
// Block: 256 threads = 8 waves (4 M x 2 N); block tile 128x128, wave tile 32x64.
// A/B 128x32 tiles double-buffered in LDS via async global->LDS DMA.
template <int EPI>
__global__ __launch_bounds__(256) void gemm_bf16_lds(
    const bf16_t* __restrict__ A, const bf16_t* __restrict__ W,
    const float* __restrict__ bias, const float* __restrict__ res,
    void* __restrict__ Cout, int M, int N, int K, int seqT) {
  __shared__ __align__(32) bf16_t smem[2][2 * 128 * 32];  // [buf][A 8KB | B 8KB]

  const int t     = threadIdx.x;
  const int lane  = t & 31;
  const int wave  = t >> 5;
  const int waveM = wave >> 1;  // 0..3
  const int waveN = wave & 1;   // 0..1
  const int bm0 = blockIdx.y * 128;
  const int bn0 = blockIdx.x * 128;
  const int m0  = bm0 + waveM * 32;
  const int n0  = bn0 + waveN * 64;

  const bf16_t* Ablk = A + (size_t)bm0 * K;
  const bf16_t* Wblk = W + (size_t)bn0 * K;

  // Copy one 128x32 A tile + one 128x32 B tile (16KB) into buffer `buf`.
  // 512 chunks of 16B each per tile; 256 threads x 2 chunks.
  auto issue = [&](int k0, int buf) {
    const unsigned la = (unsigned)(uintptr_t)&smem[buf][0];
    const unsigned lb = (unsigned)(uintptr_t)&smem[buf][128 * 32];
#pragma unroll
    for (int c = 0; c < 2; ++c) {
      const int ci  = t + c * 256;
      const int row = ci >> 2;        // 0..127
      const int e8  = (ci & 3) * 8;   // element offset within 32-wide k slab
      async_copy16(la + ci * 16, Ablk + (size_t)row * K + k0 + e8);
      async_copy16(lb + ci * 16, Wblk + (size_t)row * K + k0 + e8);
    }
  };

  v8f acc[2][4];
#pragma unroll
  for (int i = 0; i < 2; ++i)
#pragma unroll
    for (int j = 0; j < 4; ++j) acc[i][j] = (v8f){};

  issue(0, 0);
  const int ksteps = K >> 5;
  for (int kt = 0; kt < ksteps; ++kt) {
    const int buf = kt & 1;
    wait_async_all();     // my async transfers into smem[buf] are done
    __syncthreads();      // everyone's transfers are done
    if (kt + 1 < ksteps) issue((kt + 1) << 5, buf ^ 1);  // overlap next tile

    const bf16_t* la = &smem[buf][0];
    const bf16_t* lb = &smem[buf][128 * 32];
    const v16bf a0 = load_a_frag(la + (size_t)(waveM * 32 + 0) * 32, 32);
    const v16bf a1 = load_a_frag(la + (size_t)(waveM * 32 + 16) * 32, 32);
    const v16bf b0 = load_b_frag(lb + (size_t)(waveN * 64 + 0) * 32, 32);
    const v16bf b1 = load_b_frag(lb + (size_t)(waveN * 64 + 16) * 32, 32);
    const v16bf b2 = load_b_frag(lb + (size_t)(waveN * 64 + 32) * 32, 32);
    const v16bf b3 = load_b_frag(lb + (size_t)(waveN * 64 + 48) * 32, 32);
    acc[0][0] = wmma_bf16f32(a0, b0, acc[0][0]);
    acc[0][1] = wmma_bf16f32(a0, b1, acc[0][1]);
    acc[0][2] = wmma_bf16f32(a0, b2, acc[0][2]);
    acc[0][3] = wmma_bf16f32(a0, b3, acc[0][3]);
    acc[1][0] = wmma_bf16f32(a1, b0, acc[1][0]);
    acc[1][1] = wmma_bf16f32(a1, b1, acc[1][1]);
    acc[1][2] = wmma_bf16f32(a1, b2, acc[1][2]);
    acc[1][3] = wmma_bf16f32(a1, b3, acc[1][3]);
    __syncthreads();      // all waves done reading smem[buf] before overwrite
  }

  const int cl = lane & 15;
  const int rh = (lane >> 4) << 3;  // upper lane-half holds rows +8

  auto epi = [&](const v8f& a, int i, int j) {
#pragma unroll
    for (int e = 0; e < 8; ++e) {
      const int row = m0 + i * 16 + rh + e;
      const int col = n0 + j * 16 + cl;
      float v = a[e] + bias[col];
      if (EPI == EPI_BIAS_BF16) {
        ((bf16_t*)Cout)[(size_t)row * N + col] = (bf16_t)v;
      } else if (EPI == EPI_BIAS_BF16_T) {
        const int bb = row / seqT, ss = row % seqT;
        const int hh = col >> 6, dc = col & 63;
        ((bf16_t*)Cout)[(((size_t)bb * kHeads + hh) * kDk + dc) * seqT + ss] =
            (bf16_t)v;
      } else if (EPI == EPI_BIAS_RES_F32) {
        const size_t idx = (size_t)row * N + col;
        ((float*)Cout)[idx] = v + res[idx];
      } else {  // EPI_BIAS_GELU_BF16 (exact gelu)
        float g = 0.5f * v * (1.0f + erff(v * 0.70710678118654752f));
        ((bf16_t*)Cout)[(size_t)row * N + col] = (bf16_t)g;
      }
    }
  };
#pragma unroll
  for (int i = 0; i < 2; ++i)
#pragma unroll
    for (int j = 0; j < 4; ++j) epi(acc[i][j], i, j);
}

// Causal flash attention: one wave handles 16 query rows of one (b, h).
// Q, K: (B, S, d) bf16 rm (head slice via +h*64, row stride d).
// Vt: (B, heads, dk, S) bf16 (K-contiguous for the PV matmul B operand).
__global__ __launch_bounds__(128) void attn_fwd(
    const bf16_t* __restrict__ Q, const bf16_t* __restrict__ Km,
    const bf16_t* __restrict__ Vt, bf16_t* __restrict__ O) {
  __shared__ __align__(32) bf16_t pbuf[4][16 * 32];
  const int lane = threadIdx.x & 31;
  const int wave = threadIdx.x >> 5;
  const int tile = blockIdx.x * 4 + wave;
  const int qtiles = kS / 16;  // 128
  const int qt = tile % qtiles;
  const int h  = (tile / qtiles) % kHeads;
  const int b  = tile / (qtiles * kHeads);
  const int q0 = qt * 16;

  const bf16_t* Qb = Q + ((size_t)b * kS) * kD + h * kDk;
  const bf16_t* Kb = Km + ((size_t)b * kS) * kD + h * kDk;
  const bf16_t* Vb = Vt + (((size_t)b * kHeads + h) * (size_t)kDk) * kS;

  const v16bf qa0 = load_a_frag(Qb + (size_t)q0 * kD, kD);       // dk 0..31
  const v16bf qa1 = load_a_frag(Qb + (size_t)q0 * kD + 32, kD);  // dk 32..63

  v8f o0 = {}, o1 = {}, o2 = {}, o3 = {};
  float mrow[8], lrow[8];
#pragma unroll
  for (int e = 0; e < 8; ++e) { mrow[e] = -1e30f; lrow[e] = 0.0f; }

  const int cl = lane & 15;
  const int rh = (lane >> 4) << 3;
  const float scale = 0.125f;  // 1/sqrt(64)
  bf16_t* pl = &pbuf[wave][0];

  for (int k0 = 0; k0 <= q0 + 15; k0 += 32) {
    // scores: rows q0..q0+15, cols k0..k0+31 (two 16-wide N tiles)
    v8f s0 = {}, s1 = {};
    s0 = wmma_bf16f32(qa0, load_b_frag(Kb + (size_t)k0 * kD, kD), s0);
    s0 = wmma_bf16f32(qa1, load_b_frag(Kb + (size_t)k0 * kD + 32, kD), s0);
    s1 = wmma_bf16f32(qa0, load_b_frag(Kb + (size_t)(k0 + 16) * kD, kD), s1);
    s1 = wmma_bf16f32(qa1, load_b_frag(Kb + (size_t)(k0 + 16) * kD + 32, kD), s1);

#pragma unroll
    for (int e = 0; e < 8; ++e) {
      const int rowg = q0 + rh + e;
      float x0 = s0[e] * scale;
      float x1 = s1[e] * scale;
      if (k0 + cl > rowg)      x0 = -1e30f;  // causal mask
      if (k0 + 16 + cl > rowg) x1 = -1e30f;
      float tm = fmaxf(x0, x1);
#pragma unroll
      for (int off = 1; off < 16; off <<= 1)
        tm = fmaxf(tm, __shfl_xor(tm, off, 32));
      const float mnew = fmaxf(mrow[e], tm);
      const float p0 = __expf(x0 - mnew);
      const float p1 = __expf(x1 - mnew);
      const float corr = __expf(mrow[e] - mnew);
      float ps = p0 + p1;
#pragma unroll
      for (int off = 1; off < 16; off <<= 1) ps += __shfl_xor(ps, off, 32);
      lrow[e] = lrow[e] * corr + ps;
      mrow[e] = mnew;
      o0[e] *= corr; o1[e] *= corr; o2[e] *= corr; o3[e] *= corr;
      pl[(rh + e) * 32 + cl]      = (bf16_t)p0;  // P tile to LDS, 16x32 rm
      pl[(rh + e) * 32 + 16 + cl] = (bf16_t)p1;
    }
    asm volatile("s_wait_dscnt 0x0" ::: "memory");  // LDS stores visible in-wave
    const v16bf pa = load_a_frag(pl, 32);           // reshape P -> A fragment
    o0 = wmma_bf16f32(pa, load_b_frag(Vb + (size_t)0  * kS + k0, kS), o0);
    o1 = wmma_bf16f32(pa, load_b_frag(Vb + (size_t)16 * kS + k0, kS), o1);
    o2 = wmma_bf16f32(pa, load_b_frag(Vb + (size_t)32 * kS + k0, kS), o2);
    o3 = wmma_bf16f32(pa, load_b_frag(Vb + (size_t)48 * kS + k0, kS), o3);
    asm volatile("s_wait_dscnt 0x0" ::: "memory");  // WAR guard before next tile
  }

  bf16_t* Ob = O + ((size_t)b * kS + q0) * kD + h * kDk;
#pragma unroll
  for (int e = 0; e < 8; ++e) {
    const float inv = 1.0f / lrow[e];
    const size_t r = (size_t)(rh + e) * kD;
    Ob[r + 0  + cl] = (bf16_t)(o0[e] * inv);
    Ob[r + 16 + cl] = (bf16_t)(o1[e] * inv);
    Ob[r + 32 + cl] = (bf16_t)(o2[e] * inv);
    Ob[r + 48 + cl] = (bf16_t)(o3[e] * inv);
  }
}

// One block (256 threads) per row of 1024 floats; writes bf16 normalized row.
__global__ __launch_bounds__(256) void layernorm_to_bf16(
    const float* __restrict__ X, const float* __restrict__ gamma,
    const float* __restrict__ beta, bf16_t* __restrict__ Out) {
  __shared__ float red[8];
  const int row = blockIdx.x;
  const int tid = threadIdx.x;
  const float* xr = X + (size_t)row * kD;
  const float4 v = reinterpret_cast<const float4*>(xr)[tid];

  float s = v.x + v.y + v.z + v.w;
#pragma unroll
  for (int off = 16; off >= 1; off >>= 1) s += __shfl_xor(s, off, 32);
  if ((tid & 31) == 0) red[tid >> 5] = s;
  __syncthreads();
  float mean = 0.0f;
#pragma unroll
  for (int i = 0; i < 8; ++i) mean += red[i];
  mean *= (1.0f / kD);
  __syncthreads();

  const float dx = v.x - mean, dy = v.y - mean, dz = v.z - mean, dw = v.w - mean;
  float s2 = dx * dx + dy * dy + dz * dz + dw * dw;
#pragma unroll
  for (int off = 16; off >= 1; off >>= 1) s2 += __shfl_xor(s2, off, 32);
  if ((tid & 31) == 0) red[tid >> 5] = s2;
  __syncthreads();
  float var = 0.0f;
#pragma unroll
  for (int i = 0; i < 8; ++i) var += red[i];
  var *= (1.0f / kD);
  const float inv = rsqrtf(var + 1e-5f);

  const float4 g4 = reinterpret_cast<const float4*>(gamma)[tid];
  const float4 b4 = reinterpret_cast<const float4*>(beta)[tid];
  bf16_t* orow = Out + (size_t)row * kD + tid * 4;
  orow[0] = (bf16_t)(dx * inv * g4.x + b4.x);
  orow[1] = (bf16_t)(dy * inv * g4.y + b4.y);
  orow[2] = (bf16_t)(dz * inv * g4.z + b4.z);
  orow[3] = (bf16_t)(dw * inv * g4.w + b4.w);
}

__global__ void f32_to_bf16(const float* __restrict__ in,
                            bf16_t* __restrict__ outp, int n) {
  const int i = blockIdx.x * blockDim.x + threadIdx.x;
  if (i < n) outp[i] = (bf16_t)in[i];
}

}  // namespace

extern "C" void kernel_launch(void* const* d_in, const int* in_sizes, int n_in,
                              void* d_out, int out_size, void* d_ws, size_t ws_size,
                              hipStream_t stream) {
  (void)in_sizes; (void)n_in; (void)out_size; (void)ws_size;
  const float* x     = (const float*)d_in[0];
  const float* ln1_g = (const float*)d_in[1];
  const float* ln1_b = (const float*)d_in[2];
  const float* ln2_g = (const float*)d_in[3];
  const float* ln2_b = (const float*)d_in[4];
  const float* wq = (const float*)d_in[5];  const float* bq = (const float*)d_in[6];
  const float* wk = (const float*)d_in[7];  const float* bk = (const float*)d_in[8];
  const float* wv = (const float*)d_in[9];  const float* bv = (const float*)d_in[10];
  const float* wo = (const float*)d_in[11]; const float* bo = (const float*)d_in[12];
  const float* w1 = (const float*)d_in[13]; const float* b1 = (const float*)d_in[14];
  const float* w2 = (const float*)d_in[15]; const float* b2 = (const float*)d_in[16];
  float* out = (float*)d_out;

  char* ws = (char*)d_ws;
  size_t off = 0;
  auto take = [&](size_t bytes) -> char* {
    char* p = ws + off;
    off += (bytes + 255) & ~(size_t)255;
    return p;
  };

  bf16_t* wqb = (bf16_t*)take((size_t)kD * kD * 2);
  bf16_t* wkb = (bf16_t*)take((size_t)kD * kD * 2);
  bf16_t* wvb = (bf16_t*)take((size_t)kD * kD * 2);
  bf16_t* wob = (bf16_t*)take((size_t)kD * kD * 2);
  bf16_t* w1b = (bf16_t*)take((size_t)kDff * kD * 2);
  bf16_t* w2b = (bf16_t*)take((size_t)kD * kDff * 2);
  bf16_t* xn  = (bf16_t*)take((size_t)kRows * kD * 2);      // reused as xn2
  char*   big = take((size_t)4 * kRows * kD * 2);           // q|k|vt|attn (64MB)
  bf16_t* qb   = (bf16_t*)big;
  bf16_t* kb   = qb  + (size_t)kRows * kD;
  bf16_t* vtb  = kb  + (size_t)kRows * kD;
  bf16_t* attn = vtb + (size_t)kRows * kD;
  bf16_t* hdn  = (bf16_t*)big;                              // aliases q|k|vt|attn
  float*  xmid = (float*)take((size_t)kRows * kD * 4);

  // --- weights fp32 -> bf16 ---
  const int nW = kD * kD, nW1 = kDff * kD;
  f32_to_bf16<<<dim3((nW + 255) / 256), dim3(256), 0, stream>>>(wq, wqb, nW);
  f32_to_bf16<<<dim3((nW + 255) / 256), dim3(256), 0, stream>>>(wk, wkb, nW);
  f32_to_bf16<<<dim3((nW + 255) / 256), dim3(256), 0, stream>>>(wv, wvb, nW);
  f32_to_bf16<<<dim3((nW + 255) / 256), dim3(256), 0, stream>>>(wo, wob, nW);
  f32_to_bf16<<<dim3((nW1 + 255) / 256), dim3(256), 0, stream>>>(w1, w1b, nW1);
  f32_to_bf16<<<dim3((nW1 + 255) / 256), dim3(256), 0, stream>>>(w2, w2b, nW1);

  // --- attention (pre-norm) ---
  layernorm_to_bf16<<<dim3(kRows), dim3(256), 0, stream>>>(x, ln1_g, ln1_b, xn);

  const dim3 gblk(256);
  const dim3 ggrid(kD / 128, kRows / 128);  // (8, 64)
  gemm_bf16_lds<EPI_BIAS_BF16><<<ggrid, gblk, 0, stream>>>(
      xn, wqb, bq, nullptr, qb, kRows, kD, kD, kS);
  gemm_bf16_lds<EPI_BIAS_BF16><<<ggrid, gblk, 0, stream>>>(
      xn, wkb, bk, nullptr, kb, kRows, kD, kD, kS);
  gemm_bf16_lds<EPI_BIAS_BF16_T><<<ggrid, gblk, 0, stream>>>(
      xn, wvb, bv, nullptr, vtb, kRows, kD, kD, kS);

  attn_fwd<<<dim3(kB * kHeads * (kS / 16) / 4), dim3(128), 0, stream>>>(
      qb, kb, vtb, attn);

  gemm_bf16_lds<EPI_BIAS_RES_F32><<<ggrid, gblk, 0, stream>>>(
      attn, wob, bo, x, xmid, kRows, kD, kD, kS);

  // --- feed-forward (pre-norm) ---
  layernorm_to_bf16<<<dim3(kRows), dim3(256), 0, stream>>>(xmid, ln2_g, ln2_b, xn);

  const dim3 fgrid1(kDff / 128, kRows / 128);  // (32, 64)
  gemm_bf16_lds<EPI_BIAS_GELU_BF16><<<fgrid1, gblk, 0, stream>>>(
      xn, w1b, b1, nullptr, hdn, kRows, kDff, kD, kS);
  gemm_bf16_lds<EPI_BIAS_RES_F32><<<ggrid, gblk, 0, stream>>>(
      hdn, w2b, b2, xmid, out, kRows, kD, kDff, kS);
}
